// MultiAspectRetrieval_46780783788087
// MI455X (gfx1250) — matrix operand order repaired and checked by reading, not compiled, not run on hardware
//
#include <hip/hip_runtime.h>

// ---------------------------------------------------------------------------
// MultiAspectRetrieval for MI455X (gfx1250, wave32, WMMA).
//
// Dominant cost: sim GEMM  [256 x 1024] x [1024 x 100000]  (K = 8 aspects*128)
//   52.4 GFLOP over 409.6 MB of pool_keys -> HBM-bound (~17.6us @ 23.3 TB/s)
//   only if the matrix math runs at ~3 PFLOP/s -> bf16 WMMA (f32 accumulate).
//   Key L2-normalization and aspect softmax weight are per-(s,n) scalars,
//   fused into the B-tile before LDS so a single accumulator spans K=1024
//   (single pass over pool_keys).
//
// CDNA5 paths used: v_wmma_f32_16x16x32_bf16, global_load_async_to_lds_b128
// (+ s_wait_asynccnt), global_prefetch_b8, split dep-counter waits.
// ---------------------------------------------------------------------------

typedef __attribute__((ext_vector_type(16))) __bf16 v16bf;
typedef __attribute__((ext_vector_type(8)))  __bf16 v8bf;
typedef __attribute__((ext_vector_type(8)))  float  v8f;

#define S_ASPECTS 8
#define D_K       128
#define D_A       1024
#define BATCH     256
#define NPOOL     100000
#define NPAD      100032          // 1563 * 64
#define KMAX      32
#define TEMP      0.1f
#define EPSF      1e-8f
#define LDK       136             // padded LDS row stride (bf16 elems)

// -------- CDNA5 async global->LDS copy helpers ------------------------------
__device__ __forceinline__ void async_load_lds_b128(unsigned lds_off,
                                                    unsigned long long gaddr) {
  // GV mode: VDST = per-lane LDS byte offset, VADDR = 64-bit global address.
  asm volatile("global_load_async_to_lds_b128 %0, %1, off"
               :: "v"(lds_off), "v"(gaddr) : "memory");
}

__device__ __forceinline__ void wait_asynccnt0() {
#if __has_builtin(__builtin_amdgcn_s_wait_asynccnt)
  __builtin_amdgcn_s_wait_asynccnt(0);
#else
  asm volatile("s_wait_asynccnt 0x0" ::: "memory");
#endif
}

// ---------------------------------------------------------------------------
// Kernel 1: queries = einsum('ska,ba->bsk'), L2-normalized, stored bf16 as
// qn[s][b][k] (GEMM A-operand layout). One block per batch row b.
// ---------------------------------------------------------------------------
__global__ __launch_bounds__(256) void maq_queries(
    const float* __restrict__ z, const float* __restrict__ wq,
    __bf16* __restrict__ qn)
{
  __shared__ float zs[D_A];
  __shared__ float qs[S_ASPECTS * D_K];
  __shared__ float rn[S_ASPECTS];
  const int b = blockIdx.x;
  const int t = threadIdx.x;

  ((float4*)zs)[t] = ((const float4*)(z + (long)b * D_A))[t];
  __syncthreads();

  float qv[4];
#pragma unroll
  for (int j = 0; j < 4; ++j) {
    const int o = 4 * t + j;                    // o = s*128 + k
    const float* wr = wq + (long)o * D_A;       // W_Q[s][k][:] contiguous
    float acc = 0.f;
    for (int a = 0; a < D_A; a += 4) {
      float4 wv = *(const float4*)(wr + a);
      float4 zv = *(const float4*)(zs + a);
      acc += wv.x * zv.x + wv.y * zv.y + wv.z * zv.z + wv.w * zv.w;
    }
    qv[j] = acc;
    qs[o] = acc;
  }
  __syncthreads();

  if (t < S_ASPECTS) {
    float ss = 0.f;
    for (int k = 0; k < D_K; ++k) { float v = qs[t * D_K + k]; ss += v * v; }
    rn[t] = 1.f / (sqrtf(ss) + EPSF);
  }
  __syncthreads();

  const int s = (4 * t) >> 7;                   // all 4 outputs share one s
  const float r = rn[s];
#pragma unroll
  for (int j = 0; j < 4; ++j) {
    const int o = 4 * t + j;
    const int k = o & (D_K - 1);
    qn[((long)s * BATCH + b) * D_K + k] = (__bf16)(qv[j] * r);
  }
}

// ---------------------------------------------------------------------------
// Kernel 2: s_i[b,n] = sum_s w_s * (qn_s . k_{s,n}) / (||k_{s,n}|| + eps)
// Workgroup tile: M=128 (b-chunk, grid.y) x N=64 (grid.x), K=1024.
// 8 waves, each owns a 32x32 C patch = 2x2 tiles of v_wmma_f32_16x16x32_bf16.
// A tile copied global->LDS with async-to-LDS DMA; key norm + softmax weight
// folded into the B tile on the VGPR path.
// ---------------------------------------------------------------------------
__global__ __launch_bounds__(256) void maq_sim(
    const __bf16* __restrict__ qn, const float* __restrict__ keys,
    const float* __restrict__ aw, float* __restrict__ si)
{
  __shared__ __bf16 At[128][LDK];   // 34,816 B
  __shared__ __bf16 Bt[64][LDK];    // 17,408 B

  const int t    = threadIdx.x;
  const int lane = t & 31;
  const int wave = t >> 5;
  const int wm   = wave & 3;        // m base = wm*32
  const int wn   = wave >> 2;       // n base = wn*32
  const long n0  = (long)blockIdx.x * 64;
  const int  b0  = blockIdx.y * 128;

  // softmax over 8 aspect weights (scalar-ish, per thread)
  float awm = aw[0];
  for (int s2 = 1; s2 < S_ASPECTS; ++s2) awm = fmaxf(awm, aw[s2]);
  float wsum = 0.f;
  for (int s2 = 0; s2 < S_ASPECTS; ++s2) wsum += __expf(aw[s2] - awm);
  const float winv = 1.f / wsum;

  const v8f zero8 = {0.f, 0.f, 0.f, 0.f, 0.f, 0.f, 0.f, 0.f};
  v8f acc[2][2];
#pragma unroll
  for (int a2 = 0; a2 < 2; ++a2)
#pragma unroll
    for (int b2 = 0; b2 < 2; ++b2) acc[a2][b2] = zero8;

  const int  krow   = t >> 2;             // 0..63  key row within tile
  const int  kc     = (t & 3) * 32;       // 32-float chunk along k
  const long ng     = n0 + krow;
  const bool kvalid = (ng < NPOOL);
  const int  arow   = t >> 1;             // 0..127 A row
  const int  acol   = (t & 1) * 64;       // 64 bf16 per thread

  // per-thread async-copy endpoints for the A tile (contiguous 128 B)
  const unsigned at_lds  = (unsigned)(unsigned long long)(&At[arow][acol]);
  const __bf16*  at_gbl0 = qn + ((long)(b0 + arow)) * D_K + acol;

  const int mrow = lane & 15;
  const int akh  = (lane < 16) ? 0 : 8;   // A frag: K half per ISA layout
  const int bkh  = (lane < 16) ? 0 : 8;   // B frag: K half per ISA layout

#pragma unroll 1
  for (int s = 0; s < S_ASPECTS; ++s) {
    // --- prefetch next aspect's key chunk into cache (global_prefetch_b8) ---
    const float* kp = keys + ((long)s * NPOOL + (kvalid ? ng : 0)) * D_K + kc;
    if (s + 1 < S_ASPECTS && kvalid)
      __builtin_prefetch(kp + (long)NPOOL * D_K, 0, 1);

    // --- global key loads (VGPR path; need norm/scale before LDS) ---
    float4 kb[8];
    float part = 0.f;
#pragma unroll
    for (int i = 0; i < 8; ++i) {
      float4 v = kvalid ? ((const float4*)kp)[i] : make_float4(0.f, 0.f, 0.f, 0.f);
      kb[i] = v;
      part += v.x * v.x + v.y * v.y + v.z * v.z + v.w * v.w;
    }
    // deterministic row sum-of-squares across the 4 lanes that own this row
    part += __shfl_xor(part, 1);
    part += __shfl_xor(part, 2);
    const float wsc = __expf(aw[s] - awm) * winv / (sqrtf(part) + EPSF);

    __syncthreads();  // previous iteration's LDS reads complete

    // --- A tile: async DMA global->LDS (ASYNCcnt), 8 x 16 B per thread ---
    {
      unsigned long long ga =
          (unsigned long long)(at_gbl0 + (long)s * BATCH * D_K);
#pragma unroll
      for (int i = 0; i < 8; ++i)
        async_load_lds_b128(at_lds + i * 16u, ga + i * 16ull);
    }

    // --- B tile: scale + convert + store to LDS (DScnt) ---
#pragma unroll
    for (int i = 0; i < 8; ++i) {
      float4 v = kb[i];
      __bf16* bp = &Bt[krow][kc + i * 4];
      bp[0] = (__bf16)(v.x * wsc);
      bp[1] = (__bf16)(v.y * wsc);
      bp[2] = (__bf16)(v.z * wsc);
      bp[3] = (__bf16)(v.w * wsc);
    }

    wait_asynccnt0();   // barrier below only waits DScnt; ASYNCcnt is manual
    __syncthreads();    // tiles ready

    // --- WMMA over K=128 for this aspect (4 chunks of K=32) ---
#pragma unroll
    for (int kk = 0; kk < 4; ++kk) {
      union { v16bf v; v8bf h[2]; } afr[2], bfr[2];
#pragma unroll
      for (int mt = 0; mt < 2; ++mt) {
        const __bf16* p = &At[wm * 32 + mt * 16 + mrow][kk * 32 + akh];
        afr[mt].h[0] = *(const v8bf*)p;          // K = akh .. akh+7
        afr[mt].h[1] = *(const v8bf*)(p + 16);   // K = akh+16 .. akh+23
      }
#pragma unroll
      for (int nt = 0; nt < 2; ++nt) {
        const __bf16* p = &Bt[wn * 32 + nt * 16 + mrow][kk * 32 + bkh];
        bfr[nt].h[0] = *(const v8bf*)p;          // K = bkh .. bkh+7
        bfr[nt].h[1] = *(const v8bf*)(p + 16);   // K = bkh+16 .. bkh+23
      }
#pragma unroll
      for (int mt = 0; mt < 2; ++mt)
#pragma unroll
        for (int nt = 0; nt < 2; ++nt)
          acc[mt][nt] = __builtin_amdgcn_wmma_f32_16x16x32_bf16(
              false, afr[mt].v, false, bfr[nt].v, (short)0, acc[mt][nt],
              false, false);
    }
  }

  // C layout: VGPR i -> M = i + (lane<16 ? 0 : 8), N = lane & 15
#pragma unroll
  for (int mt = 0; mt < 2; ++mt) {
#pragma unroll
    for (int nt = 0; nt < 2; ++nt) {
      const int  r0 = b0 + wm * 32 + mt * 16 + ((lane < 16) ? 0 : 8);
      const long c  = n0 + wn * 32 + nt * 16 + (lane & 15);
#pragma unroll
      for (int i = 0; i < 8; ++i)
        si[(long)(r0 + i) * NPAD + c] = acc[mt][nt][i];
    }
  }
}

// ---------------------------------------------------------------------------
// Kernel 3: per-row top-32 + gated/warmup alpha. One block per b.
// Keys are (sortable_f32_bits << 32) | ~index  -> ties pick lowest index.
// Per-thread sorted lists in LDS (early reject), then 8 pairwise merge rounds.
// ---------------------------------------------------------------------------
__global__ __launch_bounds__(256) void maq_topk(
    const float* __restrict__ si, const float* __restrict__ tau_p,
    const float* __restrict__ lam_p, const unsigned char* __restrict__ warm_p,
    float* __restrict__ out_alpha, int* __restrict__ out_idx)
{
  extern __shared__ unsigned long long sm[];
  unsigned long long* buf0 = sm;              // 256 lists, stride 33
  unsigned long long* buf1 = sm + 256 * 33;
  const int b = blockIdx.x;
  const int t = threadIdx.x;
  const int base = t * 33;

#pragma unroll
  for (int j = 0; j < 32; ++j) buf0[base + j] = 0ull;
  unsigned long long lmin = 0ull;

  const float* row = si + (long)b * NPAD;
  for (int n = t; n < NPOOL; n += 256) {
    float v = row[n];
    unsigned u = __float_as_uint(v);
    u = (u & 0x80000000u) ? ~u : (u | 0x80000000u);
    unsigned long long key =
        ((unsigned long long)u << 32) | (unsigned)(~(unsigned)n);
    if (key > lmin) {
      int j = 31;
      while (j > 0 && buf0[base + j - 1] < key) {
        buf0[base + j] = buf0[base + j - 1];
        --j;
      }
      buf0[base + j] = key;
      lmin = buf0[base + 31];
    }
  }
  __syncthreads();

  unsigned long long* src = buf0;
  unsigned long long* dst = buf1;
  for (int half = 128; half >= 1; half >>= 1) {
    if (t < half) {
      const unsigned long long* la = src + (long)t * 33;
      const unsigned long long* lb = src + (long)(t + half) * 33;
      unsigned long long* lo = dst + (long)t * 33;
      int i = 0, j = 0;
      unsigned long long a = la[0], c = lb[0];
#pragma unroll 1
      for (int p = 0; p < 32; ++p) {
        if (a >= c) { lo[p] = a; ++i; a = (i < 32) ? la[i] : 0ull; }
        else        { lo[p] = c; ++j; c = (j < 32) ? lb[j] : 0ull; }
      }
    }
    __syncthreads();
    unsigned long long* tmp = src; src = dst; dst = tmp;
  }
  // sorted-descending top-32 now in src[0..31]; dst region is free scratch

  const bool  warm = (warm_p[0] != 0);
  const float tau  = tau_p[0];
  const float lam  = lam_p[0];
  float* red = (float*)dst;
  float e = 0.f;
  int   idx = 0;
  if (t < KMAX) {
    unsigned long long key = src[t];
    unsigned eu = (unsigned)(key >> 32);
    unsigned uo = (eu & 0x80000000u) ? (eu ^ 0x80000000u) : ~eu;
    float v = __uint_as_float(uo);
    idx = (int)(~(unsigned)key);

    unsigned long long k0 = src[0];
    unsigned e0 = (unsigned)(k0 >> 32);
    unsigned u0 = (e0 & 0x80000000u) ? (e0 ^ 0x80000000u) : ~e0;
    float vmax = __uint_as_float(u0);

    if (warm) {
      e = __expf((v - vmax) / TEMP);                    // softmax(scores/T)
    } else {
      float g = 1.f / (1.f + __expf(-lam * (v - tau))); // sigmoid gate
      e = g * __expf((v - vmax) / TEMP);                // norms cancel
    }
    red[t] = e;
  }
  __syncthreads();
  if (t < KMAX) {
    float sum = 0.f;
    for (int j = 0; j < KMAX; ++j) sum += red[j];
    out_alpha[(long)b * KMAX + t] = e / (sum + EPSF);
    out_idx[(long)b * KMAX + t]   = idx;
  }
}

// ---------------------------------------------------------------------------
extern "C" void kernel_launch(void* const* d_in, const int* in_sizes, int n_in,
                              void* d_out, int out_size, void* d_ws,
                              size_t ws_size, hipStream_t stream)
{
  const float* z          = (const float*)d_in[0];
  const float* pool_keys  = (const float*)d_in[1];
  const float* wq         = (const float*)d_in[2];
  const float* aw         = (const float*)d_in[3];
  const float* tau        = (const float*)d_in[4];
  const float* lam        = (const float*)d_in[5];
  const unsigned char* wp = (const unsigned char*)d_in[6];

  // workspace: qn bf16 [8][256][128] (512 KB) then s_i f32 [256][100032] (~102 MB)
  __bf16* qn = (__bf16*)d_ws;
  float*  si = (float*)((char*)d_ws + (size_t)S_ASPECTS * BATCH * D_K * 2);

  // outputs: alphas f32 [256][32] then indices i32 [256][32] (raw int32 bits)
  float* out_alpha = (float*)d_out;
  int*   out_idx   = (int*)d_out + BATCH * KMAX;

  maq_queries<<<BATCH, 256, 0, stream>>>(z, wq, qn);

  dim3 g2((NPOOL + 63) / 64, BATCH / 128);
  maq_sim<<<g2, 256, 0, stream>>>(qn, pool_keys, aw, si);

  size_t smem = 2ull * 256 * 33 * sizeof(unsigned long long);  // 135,168 B
  maq_topk<<<BATCH, 256, smem, stream>>>(si, tau, lam, wp, out_alpha, out_idx);
}
